// ConvexLORALinear_85633057948172
// MI455X (gfx1250) — matrix operand order repaired and checked by reading, not compiled, not run on hardware
//
#include <hip/hip_runtime.h>
#include <stdint.h>

typedef __attribute__((ext_vector_type(16))) __bf16 v16bf;
typedef __attribute__((ext_vector_type(8)))  __bf16 bf16x8;
typedef __attribute__((ext_vector_type(2)))  __bf16 bf16x2;
typedef __attribute__((ext_vector_type(4)))  __bf16 bf16x4;
typedef __attribute__((ext_vector_type(8)))  float  v8f;
typedef __attribute__((ext_vector_type(4)))  float  f32x4;
typedef __attribute__((ext_vector_type(2)))  float  f32x2;

// ---------------------------------------------------------------------------
// Pre-pass: convert f32 B[K,N] (row-major) to bf16, swizzled into WMMA
// B-fragment lane order. For K-tile kt and N-tile nt (tile = 32x16), lane l
// needs 16 bf16: column n = nt*16 + (l&15), K = kt*32 + (l>>4)*8 + {0..7, 16..23}.
// Stored contiguously: out[(kt*(N/16)+nt)*512 + l*16 + e]  -> a wave later
// reads a whole fragment as 1KB of contiguous memory (2x b128 per lane).
// One thread per (tile, lane): total threads = (K/32)*(N/16)*32 = K*N/16.
// ---------------------------------------------------------------------------
template <int K, int N>
__global__ __launch_bounds__(256)
void swizzle_b_bf16(const float* __restrict__ B, __bf16* __restrict__ out) {
  const uint32_t t    = blockIdx.x * 256u + threadIdx.x;
  const uint32_t lane = t & 31u;
  const uint32_t ti   = t >> 5;            // tile index = kt*(N/16) + nt
  constexpr uint32_t N16 = N / 16;
  const uint32_t kt   = ti / N16;          // power-of-two divide -> shift
  const uint32_t nt   = ti % N16;
  const uint32_t half = lane >> 4;
  const uint32_t l16  = lane & 15u;

  const uint32_t n     = nt * 16u + l16;
  const uint32_t kbase = kt * 32u + half * 8u;

  bf16x8 lo, hi;
  #pragma unroll
  for (uint32_t v = 0; v < 4; ++v) {
    f32x2 t0, t1;
    t0[0] = B[(size_t)(kbase + 2u * v)      * N + n];
    t0[1] = B[(size_t)(kbase + 2u * v + 1u) * N + n];
    t1[0] = B[(size_t)(kbase + 16u + 2u * v)      * N + n];
    t1[1] = B[(size_t)(kbase + 16u + 2u * v + 1u) * N + n];
    bf16x2 b0 = __builtin_convertvector(t0, bf16x2);
    bf16x2 b1 = __builtin_convertvector(t1, bf16x2);
    lo[2 * v] = b0[0]; lo[2 * v + 1] = b0[1];
    hi[2 * v] = b1[0]; hi[2 * v + 1] = b1[1];
  }
  __bf16* dst = out + (size_t)ti * 512u + lane * 16u;
  *(bf16x8*)&dst[0] = lo;   // global_store_b128
  *(bf16x8*)&dst[8] = hi;   // global_store_b128
}

// ---------------------------------------------------------------------------
// GEMM  C[M,N] = A[M,K] * Bsw  (Bsw pre-swizzled bf16 fragments).
// TA: float (convert on the fly) or __bf16 (direct). TC: float or __bf16.
// Block = 256 threads = 8 wave32s arranged 4 row-slabs x 2 wave-columns.
// Wave (mw,nw) owns rows [blockIdx.y*64 + mw*16, +16) and TWO 16-col tiles
// at cols [blockIdx.x*64 + nw*32, +32): one A fragment feeds two WMMAs.
// Grid covers M,N exactly -> EXEC all-ones around every WMMA (ISA 7.12).
// ---------------------------------------------------------------------------
template <int LDA, int N16, int KTOT, int LDC, typename TA, typename TC>
__global__ __launch_bounds__(256)
void lora_gemm_bf16_wmma(const TA* __restrict__ A,
                         const __bf16* __restrict__ Bsw,
                         TC* __restrict__ C) {
  const uint32_t lane = threadIdx.x & 31u;
  const uint32_t wave = threadIdx.x >> 5;
  const uint32_t mw   = wave >> 1;    // 0..3 row-slab within block
  const uint32_t nw   = wave & 1u;    // 0..1 column group within block
  const uint32_t half = lane >> 4;    // 0: lanes 0-15, 1: lanes 16-31
  const uint32_t l16  = lane & 15u;

  const uint32_t rowBase   = blockIdx.y * 64u + mw * 16u;
  const uint32_t nTileBase = blockIdx.x * 4u + nw * 2u;   // 2 tiles per wave
  const uint32_t colBase   = nTileBase * 16u;

  // A fragment (16x32 bf16, ISA 7.12.2): lane l -> row l&15;
  // lanes 0-15: K=[0..7],[16..23]; lanes 16-31: K=[8..15],[24..31].
  const TA* Alane = A + (size_t)rowBase * LDA + l16 * LDA + half * 8u;

  // B fragments: two consecutive pre-swizzled 1KB tiles.
  const __bf16* Bw = Bsw + (size_t)nTileBase * 512u + lane * 16u;

  v8f acc0 = {};
  v8f acc1 = {};

  #pragma unroll 2
  for (int k0 = 0; k0 < KTOT; k0 += 32) {
    __builtin_prefetch(Alane + 32, 0, 1);   // next A K-slab

    // ---- A fragment (shared by both output tiles) ----
    v16bf afrag;
    if constexpr (sizeof(TA) == 4) {
      f32x4 a0 = *(const f32x4*)&Alane[0];
      f32x4 a1 = *(const f32x4*)&Alane[4];
      f32x4 a2 = *(const f32x4*)&Alane[16];
      f32x4 a3 = *(const f32x4*)&Alane[20];
      bf16x4 c0 = __builtin_convertvector(a0, bf16x4);
      bf16x4 c1 = __builtin_convertvector(a1, bf16x4);
      bf16x4 c2 = __builtin_convertvector(a2, bf16x4);
      bf16x4 c3 = __builtin_convertvector(a3, bf16x4);
      #pragma unroll
      for (int i = 0; i < 4; ++i) {
        afrag[i]      = c0[i];
        afrag[4 + i]  = c1[i];
        afrag[8 + i]  = c2[i];
        afrag[12 + i] = c3[i];
      }
    } else {
      bf16x8 r0 = *(const bf16x8*)&Alane[0];    // global_load_b128
      bf16x8 r1 = *(const bf16x8*)&Alane[16];   // global_load_b128
      #pragma unroll
      for (int i = 0; i < 8; ++i) {
        afrag[i]     = r0[i];
        afrag[8 + i] = r1[i];
      }
    }

    // ---- B fragments: 2 tiles x 2 coalesced b128 (immediate offsets) ----
    bf16x8 b00 = *(const bf16x8*)&Bw[0];
    bf16x8 b01 = *(const bf16x8*)&Bw[8];
    bf16x8 b10 = *(const bf16x8*)&Bw[512];
    bf16x8 b11 = *(const bf16x8*)&Bw[520];
    v16bf bfrag0, bfrag1;
    #pragma unroll
    for (int i = 0; i < 8; ++i) {
      bfrag0[i]     = b00[i];
      bfrag0[8 + i] = b01[i];
      bfrag1[i]     = b10[i];
      bfrag1[8 + i] = b11[i];
    }

    // Two D = A*B + C per K-step (f32 accumulate); A fragment reused.
    acc0 = __builtin_amdgcn_wmma_f32_16x16x32_bf16(
        false, afrag, false, bfrag0, (short)0, acc0, false, false);
    acc1 = __builtin_amdgcn_wmma_f32_16x16x32_bf16(
        false, afrag, false, bfrag1, (short)0, acc1, false, false);

    Alane += 32;                    // one per-lane bump
    Bw    += (size_t)N16 * 512u;    // next K-tile row of fragments
  }

  // ---- Store C (16x16 f32): lanes 0-15 -> M=vgpr, lanes 16-31 -> M=vgpr+8 ----
  TC* Clane = C + (size_t)(rowBase + half * 8u) * LDC + colBase + l16;
  #pragma unroll
  for (int v = 0; v < 8; ++v) {
    Clane[(size_t)v * LDC]       = (TC)acc0[v];
    Clane[(size_t)v * LDC + 16u] = (TC)acc1[v];
  }
}

extern "C" void kernel_launch(void* const* d_in, const int* in_sizes, int n_in,
                              void* d_out, int out_size, void* d_ws, size_t ws_size,
                              hipStream_t stream) {
  (void)in_sizes; (void)n_in; (void)out_size; (void)ws_size;

  const float* input = (const float*)d_in[0];  // [8192, 4096]
  const float* loraA = (const float*)d_in[1];  // [4096, 128]
  const float* loraB = (const float*)d_in[2];  // [128, 4096]
  float* out = (float*)d_out;                  // [8192, 4096]

  constexpr int TOKENS  = 8192;
  constexpr int IN_DIM  = 4096;
  constexpr int OUT_DIM = 4096;
  constexpr int RANK    = 128;

  // Workspace layout (4 MB total):
  //   [0, 1MB)   swizzled bf16 lora_A fragments
  //   [1, 2MB)   swizzled bf16 lora_B fragments
  //   [2, 4MB)   bf16 intermediate T[8192,128]
  char* ws = (char*)d_ws;
  __bf16* swA = (__bf16*)(ws);
  __bf16* swB = (__bf16*)(ws + (size_t)IN_DIM * RANK * 2);
  __bf16* tmp = (__bf16*)(ws + (size_t)IN_DIM * RANK * 2 + (size_t)RANK * OUT_DIM * 2);

  // Pre-swizzle both LoRA factors (one thread per tile-lane; K*N/16 threads).
  swizzle_b_bf16<IN_DIM, RANK>
      <<<dim3(IN_DIM * RANK / 16 / 256), 256, 0, stream>>>(loraA, swA);
  swizzle_b_bf16<RANK, OUT_DIM>
      <<<dim3(RANK * OUT_DIM / 16 / 256), 256, 0, stream>>>(loraB, swB);

  // Stage 1: T = input @ lora_A   (M=8192, N=128, K=4096), T stored bf16.
  // Block tile 64x64 -> grid (128/64, 8192/64) = (2, 128).
  lora_gemm_bf16_wmma<IN_DIM, RANK / 16, IN_DIM, RANK, float, __bf16>
      <<<dim3(RANK / 64, TOKENS / 64), 256, 0, stream>>>(input, swA, tmp);

  // Stage 2: out = T @ lora_B     (M=8192, N=4096, K=128), out f32.
  // Block tile 64x64 -> grid (4096/64, 8192/64) = (64, 128).
  lora_gemm_bf16_wmma<RANK, OUT_DIM / 16, RANK, OUT_DIM, __bf16, float>
      <<<dim3(OUT_DIM / 64, TOKENS / 64), 256, 0, stream>>>(tmp, swB, out);
}
